// Int8Linear_10857677324715
// MI455X (gfx1250) — matrix operand order repaired and checked by reading, not compiled, not run on hardware
//
#include <hip/hip_runtime.h>
#include <stdint.h>

typedef __attribute__((ext_vector_type(16))) _Float16 v16h;
typedef __attribute__((ext_vector_type(8)))  _Float16 v8h;
typedef __attribute__((ext_vector_type(2)))  _Float16 v2h;
typedef __attribute__((ext_vector_type(8)))  float    v8f;
typedef __attribute__((ext_vector_type(4)))  int      v4i;

#define K_DIM 4096
#define N_DIM 11008
#define M_DIM 8192       // B*S = 4*2048
#define KSTEP 64         // K depth per LDS stage (2 WMMA sub-steps)
#define NSTAGES (K_DIM / KSTEP)   // 64
#define A_STRIDE 144     // 128B row data + 16B pad: ds_load_b128 conflict-free
#define B_STRIDE 80      // 64B col data + 16B pad: ds_load_b128 conflict-free

// Exact int8 -> fp16 dequant of 16 bytes using v_perm_b32 + v_pk_add_f16.
//   xu = b ^ 0x80;  bits 0x6400|xu == fp16(1024+xu) exactly;  -1152 recovers b.
__device__ __forceinline__ v16h dequant16(v4i raw) {
  union { v16h v; v2h p[8]; } r;
  const uint32_t magic = 0x64646464u;
  const v2h off = { (_Float16)1152.0f, (_Float16)1152.0f };
#pragma unroll
  for (int d = 0; d < 4; ++d) {
    uint32_t wx = ((uint32_t)raw[d]) ^ 0x80808080u;
    uint32_t lo = __builtin_amdgcn_perm(magic, wx, 0x04010400u);
    uint32_t hi = __builtin_amdgcn_perm(magic, wx, 0x04030402u);
    r.p[2 * d + 0] = __builtin_bit_cast(v2h, lo) - off;
    r.p[2 * d + 1] = __builtin_bit_cast(v2h, hi) - off;
  }
  return r.v;
}

// One 16-byte async global->LDS copy (per-lane), tracked by ASYNCcnt.
__device__ __forceinline__ void async_copy16(uint32_t lds_addr, const void* gptr) {
  asm volatile("global_load_async_to_lds_b128 %0, %1, off"
               :: "v"(lds_addr), "v"((uint64_t)(uintptr_t)gptr) : "memory");
}

// Block tile: 128(M) x 128(N), 256 threads = 8 waves in 2x4 (wm x wn).
// Wave tile:  64(M) x 32(N) = 4x2 WMMA 16x16 tiles.
// K loop: 64 stages of 64 (2 sub-steps of 32), 4-deep LDS async pipeline.
__global__ __launch_bounds__(256) void int8_linear_wmma_kernel(
    const _Float16* __restrict__ X,   // [M_DIM, K_DIM] fp16, K-major
    const int8_t*   __restrict__ W,   // [N_DIM, K_DIM] int8, K-major
    const float*    __restrict__ S,   // [N_DIM] fp32 per-channel scales
    _Float16*       __restrict__ Y)   // [M_DIM, N_DIM] fp16
{
  __shared__ __align__(16) uint8_t ldsA[4][128 * A_STRIDE];  // 72 KB fp16 A tiles
  __shared__ __align__(16) uint8_t ldsB[4][128 * B_STRIDE];  // 40 KB raw int8 B tiles

  const int t    = threadIdx.x;
  const int lane = t & 31;
  const int wave = t >> 5;
  const int wm   = wave >> 2;          // 0..1
  const int wn   = wave & 3;           // 0..3
  const int hi   = lane >> 4;          // 0..1 (lane half, per WMMA layouts)
  const int lr   = lane & 15;          // 0..15

  const int m_blk = blockIdx.y * 128;
  const int n_blk = blockIdx.x * 128;

  // Copy split per stage: A = 1024 x 16B chunks (4/thread), B = 512 x 16B (2/thread).
  const int a_row = t >> 3, a_seg = t & 7;   // A rows a_row + {0,32,64,96}, segment a_seg
  const int b_col = t >> 2, b_seg = t & 3;   // B cols b_col + {0,64}, segment b_seg

  auto issue = [&](int s) {
    const int k0 = s * KSTEP;
    uint8_t* bufA = ldsA[s & 3];
    uint8_t* bufB = ldsB[s & 3];
#pragma unroll
    for (int u = 0; u < 4; ++u) {
      const int row = a_row + u * 32;
      async_copy16((uint32_t)(uintptr_t)(bufA + row * A_STRIDE + a_seg * 16),
                   X + (size_t)(m_blk + row) * K_DIM + k0 + a_seg * 8);
    }
#pragma unroll
    for (int u = 0; u < 2; ++u) {
      const int col = b_col + u * 64;
      async_copy16((uint32_t)(uintptr_t)(bufB + col * B_STRIDE + b_seg * 16),
                   W + (size_t)(n_blk + col) * K_DIM + k0 + b_seg * 16);
    }
  };

  v8f acc[4][2];
#pragma unroll
  for (int i = 0; i < 4; ++i)
#pragma unroll
    for (int j = 0; j < 2; ++j)
      acc[i][j] = (v8f){0.f, 0.f, 0.f, 0.f, 0.f, 0.f, 0.f, 0.f};

  // One 32-deep WMMA sub-step: hoist ALL ds loads into distinct registers
  // first (partial dscnt waits), then dequant, then 8 WMMAs.
  auto compute_sub = [&](const uint8_t* bufA, const uint8_t* bufB, int ss) {
    v4i braw[2];
#pragma unroll
    for (int j = 0; j < 2; ++j) {
      const int col = wn * 32 + j * 16 + lr;
      braw[j] = *(const v4i*)(bufB + col * B_STRIDE + ss * 32 + hi * 16);
    }
    v8h alo[4], ahi[4];
#pragma unroll
    for (int i = 0; i < 4; ++i) {
      const int row = wm * 64 + i * 16 + lr;
      alo[i] = *(const v8h*)(bufA + row * A_STRIDE + ss * 64 + hi * 32);
      ahi[i] = *(const v8h*)(bufA + row * A_STRIDE + ss * 64 + hi * 32 + 16);
    }
    v16h b[2];
#pragma unroll
    for (int j = 0; j < 2; ++j)
      b[j] = dequant16(braw[j]);
#pragma unroll
    for (int i = 0; i < 4; ++i) {
      union { v16h v; v8h h[2]; } ua;
      ua.h[0] = alo[i];
      ua.h[1] = ahi[i];
#pragma unroll
      for (int j = 0; j < 2; ++j)
        acc[i][j] = __builtin_amdgcn_wmma_f32_16x16x32_f16(
            false, ua.v, false, b[j], (short)0, acc[i][j], false, false);
    }
  };

  // ---- 4-buffer, depth-2 async pipeline; one barrier per 64-deep stage ----
  issue(0);
  issue(1);
  for (int s = 0; s < NSTAGES; ++s) {
    if (s + 2 < NSTAGES) {
      issue(s + 2);
      // outstanding = stages s+1, s+2 (6 async ops each); <=12 => stage s done
      asm volatile("s_wait_asynccnt 0xc" ::: "memory");
    } else if (s + 1 < NSTAGES) {
      asm volatile("s_wait_asynccnt 0x6" ::: "memory");
    } else {
      asm volatile("s_wait_asynccnt 0x0" ::: "memory");
    }
    __syncthreads();          // stage-s tiles visible WGP-wide; frees buf (s+3)&3 for the next issue
    const uint8_t* bufA = ldsA[s & 3];
    const uint8_t* bufB = ldsB[s & 3];
    compute_sub(bufA, bufB, 0);   // K = s*64 + [0,32)
    compute_sub(bufA, bufB, 1);   // K = s*64 + [32,64)
  }

  // ---- Epilogue: per-channel scale, cast to fp16, non-temporal stores ----
  // C/D layout: VGPR r -> row (r + 8*hi) within tile, col lr.
#pragma unroll
  for (int j = 0; j < 2; ++j) {
    const int n = n_blk + wn * 32 + j * 16 + lr;
    const float sc = S[n];
#pragma unroll
    for (int i = 0; i < 4; ++i) {
      const int mbase = m_blk + wm * 64 + i * 16 + hi * 8;
#pragma unroll
      for (int r = 0; r < 8; ++r) {
        _Float16 val = (_Float16)(acc[i][j][r] * sc);
        __builtin_nontemporal_store(val, Y + (size_t)(mbase + r) * N_DIM + n);
      }
    }
  }
}

extern "C" void kernel_launch(void* const* d_in, const int* in_sizes, int n_in,
                              void* d_out, int out_size, void* d_ws, size_t ws_size,
                              hipStream_t stream) {
  const _Float16* x = (const _Float16*)d_in[0];   // fp16 activations
  const int8_t*   w = (const int8_t*)d_in[1];     // int8 weights [N, K]
  const float*    s = (const float*)d_in[2];      // fp32 scales [N]
  _Float16*       y = (_Float16*)d_out;           // fp16 output [M, N]

  dim3 grid(N_DIM / 128, M_DIM / 128);            // 86 x 64 workgroups
  int8_linear_wmma_kernel<<<grid, dim3(256), 0, stream>>>(x, w, s, y);
}
